// Net_3710851743887
// MI455X (gfx1250) — compile-verified, matrix operand order checked by reading, not run on hardware
//
#include <hip/hip_runtime.h>

// ---------------------------------------------------------------------------
// Problem constants (match reference)
// ---------------------------------------------------------------------------
#define BATCH   8192
#define DIM     2048     // U_DIM == V_DIM == LATENT == 2048
#define SLOPE   0.001f
#define LN_EPS  1e-5f

#define TM 128           // M tile (batch rows)
#define TN 128           // N tile (latent cols)
#define TK 32            // K step  (bf16 WMMA K)

typedef __attribute__((ext_vector_type(16))) __bf16 v16bf;
typedef __attribute__((ext_vector_type(8)))  float  v8f;
typedef __attribute__((ext_vector_type(4)))  unsigned int u32x4;
typedef __attribute__((ext_vector_type(8)))  int    i32x8;
typedef __attribute__((ext_vector_type(4)))  int    i32x4;

union FragBF {
    v16bf v;
    uint4 q[2];
};

__device__ __forceinline__ unsigned short f32_to_bf16(float f) {
    unsigned int u = __float_as_uint(f);
    unsigned int r = 0x7FFFu + ((u >> 16) & 1u);   // round-to-nearest-even
    return (unsigned short)((u + r) >> 16);
}

__device__ __forceinline__ unsigned int pack_bf16x2(float lo, float hi) {
    return (unsigned int)f32_to_bf16(lo) | ((unsigned int)f32_to_bf16(hi) << 16);
}

__device__ __forceinline__ float leaky(float x) {
    return x >= 0.0f ? x : SLOPE * x;
}

// ---------------------------------------------------------------------------
// Kernel 0: zero the output accumulator
// ---------------------------------------------------------------------------
__global__ void zero_out_kernel(float* __restrict__ out, int n) {
    int i = blockIdx.x * blockDim.x + threadIdx.x;
    if (i < n) out[i] = 0.0f;
}

// ---------------------------------------------------------------------------
// Kernel 1: per-row LayerNorm -> bf16 (one 256-thread block per 2048-row)
// ---------------------------------------------------------------------------
__global__ __launch_bounds__(256)
void ln_rows_kernel(const float* __restrict__ x,
                    const float* __restrict__ gamma,
                    const float* __restrict__ beta,
                    unsigned short* __restrict__ y)
{
    __shared__ float ssum[256];
    __shared__ float ssq[256];

    const int tid = threadIdx.x;
    const int row = blockIdx.x;
    const float* xr = x + (size_t)row * DIM;

    const float4* xr4 = (const float4*)(xr + tid * 8);
    float4 p0 = xr4[0];
    float4 p1 = xr4[1];

    float s  = p0.x + p0.y + p0.z + p0.w + p1.x + p1.y + p1.z + p1.w;
    float sq = p0.x*p0.x + p0.y*p0.y + p0.z*p0.z + p0.w*p0.w
             + p1.x*p1.x + p1.y*p1.y + p1.z*p1.z + p1.w*p1.w;

    ssum[tid] = s;
    ssq[tid]  = sq;
    __syncthreads();
    for (int off = 128; off > 0; off >>= 1) {
        if (tid < off) {
            ssum[tid] += ssum[tid + off];
            ssq[tid]  += ssq[tid + off];
        }
        __syncthreads();
    }
    const float inv_n = 1.0f / (float)DIM;
    const float mu    = ssum[0] * inv_n;
    const float var   = ssq[0] * inv_n - mu * mu;   // biased var (matches jnp.var)
    const float rs    = rsqrtf(var + LN_EPS);

    const float4* g4 = (const float4*)(gamma + tid * 8);
    const float4* b4 = (const float4*)(beta  + tid * 8);
    float4 g0 = g4[0], g1 = g4[1];
    float4 b0 = b4[0], b1 = b4[1];

    float o[8];
    o[0] = (p0.x - mu) * rs * g0.x + b0.x;
    o[1] = (p0.y - mu) * rs * g0.y + b0.y;
    o[2] = (p0.z - mu) * rs * g0.z + b0.z;
    o[3] = (p0.w - mu) * rs * g0.w + b0.w;
    o[4] = (p1.x - mu) * rs * g1.x + b1.x;
    o[5] = (p1.y - mu) * rs * g1.y + b1.y;
    o[6] = (p1.z - mu) * rs * g1.z + b1.z;
    o[7] = (p1.w - mu) * rs * g1.w + b1.w;

    uint4 packed = make_uint4(pack_bf16x2(o[0], o[1]), pack_bf16x2(o[2], o[3]),
                              pack_bf16x2(o[4], o[5]), pack_bf16x2(o[6], o[7]));
    *(uint4*)(y + (size_t)row * DIM + tid * 8) = packed;
}

// ---------------------------------------------------------------------------
// Kernel 2: fused dual-GEMM + bias + LeakyReLU + hu.hv dot reduction
//   grid: (BATCH/TM, DIM/TN), 256 threads = 8 waves (wave32)
//   A tiles staged by the Tensor Data Mover (tensor_load_to_lds + TENSORcnt),
//   one 32x32 bf16 slice per wave. B tiles staged/converted by the waves.
//   Inner loop software-pipelined: B frags for tile j+1 load under WMMA j.
// ---------------------------------------------------------------------------
__global__ __launch_bounds__(256)
void fused_gemm_dot_kernel(const unsigned short* __restrict__ Au,   // [BATCH][DIM] bf16
                           const unsigned short* __restrict__ Av,   // [BATCH][DIM] bf16
                           const float* __restrict__ Wu,            // [DIM][DIM] f32
                           const float* __restrict__ Wv,            // [DIM][DIM] f32
                           const float* __restrict__ bu,            // [DIM] f32
                           const float* __restrict__ bv,            // [DIM] f32
                           float* __restrict__ out)                 // [BATCH] f32 (atomic acc)
{
    __shared__ unsigned short sAu[TM][TK];   // 8 KB (TDM-written)
    __shared__ unsigned short sAv[TM][TK];   // 8 KB (TDM-written)
    __shared__ unsigned short sBu[TN][TK];   // 8 KB ([n][k] transposed bf16 W tile)
    __shared__ unsigned short sBv[TN][TK];   // 8 KB
    __shared__ float          sRow[TM];

    const int tid  = threadIdx.x;
    const int lane = tid & 31;
    const int wave = tid >> 5;
    const int M0   = blockIdx.x * TM;
    const int N0   = blockIdx.y * TN;

    if (tid < TM) sRow[tid] = 0.0f;

    v8f accU[8] = {};
    v8f accV[8] = {};

    // ---- TDM slice assignment: waves 0..3 -> Au rows, waves 4..7 -> Av rows
    const int             tdm_mat = wave >> 2;           // 0 = Au, 1 = Av
    const int             tdm_row = (wave & 3) * 32;     // 32-row slice of the tile
    const unsigned short* tdm_src = tdm_mat ? Av : Au;
    unsigned short*       tdm_lds = tdm_mat ? &sAv[0][0] : &sAu[0][0];
    const int tdm_loff = (int)__builtin_amdgcn_readfirstlane(
        (int)(unsigned int)(size_t)(tdm_lds + (size_t)tdm_row * TK));

    // D# group1 (ISA 8.4): data_size=2B, tensor_dim0=DIM, tensor_dim1=BATCH,
    // tile_dim0=TK, tile_dim1=32, tensor_dim0_stride=DIM
    i32x8 g1;
    g1[0] = (int)(1u << 16);                                   // data_size = 1 -> 2 bytes
    g1[1] = (int)((DIM & 0xFFFFu) << 16);                      // tensor_dim0[15:0]
    g1[2] = (int)(((DIM >> 16) & 0xFFFFu) |
                  ((BATCH & 0xFFFFu) << 16));                  // dim0[31:16] | dim1[15:0]
    g1[3] = (int)(((BATCH >> 16) & 0xFFFFu) |
                  ((unsigned)TK << 16));                       // dim1[31:16] | tile_dim0
    g1[4] = 32;                                                // tile_dim1=32, tile_dim2=0
    g1[5] = DIM;                                               // tensor_dim0_stride[31:0]
    g1[6] = 0;
    g1[7] = 0;
    const i32x4 zg4 = {0, 0, 0, 0};
    const i32x8 zg8 = {0, 0, 0, 0, 0, 0, 0, 0};

    // ---- B-tile staging: thread converts 2 K-rows x 8 N-cols, packs pairs
    const int kp = tid >> 4;            // 0..15  (k = 2*kp)
    const int bn = (tid & 15) * 8;      // 0..120

    for (int k0 = 0; k0 < DIM; k0 += TK) {
        __syncthreads();   // previous iteration's fragment reads complete

        // ---- A tiles via Tensor Data Mover (one 32x32 bf16 slice per wave) ----
        {
            const unsigned long long ga = (unsigned long long)(size_t)
                (tdm_src + (size_t)(M0 + tdm_row) * DIM + k0);
            u32x4 g0;
            g0[0] = 1u;                                        // count=1 (valid D#)
            g0[1] = (unsigned int)tdm_loff;                    // lds_addr
            g0[2] = (unsigned int)__builtin_amdgcn_readfirstlane((int)(unsigned int)ga);
            g0[3] = ((unsigned int)__builtin_amdgcn_readfirstlane(
                         (int)(unsigned int)(ga >> 32)) & 0x01FFFFFFu)
                    | (2u << 30);                              // global_addr[56:32] | type=2
            __builtin_amdgcn_tensor_load_to_lds(g0, g1, zg4, zg4, zg8, 0);
        }

        // ---- B tiles: f32 -> bf16 convert + transpose into [n][k] ----
        {
            const int kk = k0 + 2 * kp;
            const float4* r0u = (const float4*)(Wu + (size_t)kk       * DIM + N0 + bn);
            const float4* r1u = (const float4*)(Wu + (size_t)(kk + 1) * DIM + N0 + bn);
            const float4* r0v = (const float4*)(Wv + (size_t)kk       * DIM + N0 + bn);
            const float4* r1v = (const float4*)(Wv + (size_t)(kk + 1) * DIM + N0 + bn);
            float4 a0 = r0u[0], a1 = r0u[1], c0 = r1u[0], c1 = r1u[1];
            float4 e0 = r0v[0], e1 = r0v[1], f0 = r1v[0], f1 = r1v[1];
            const float u0[8] = {a0.x,a0.y,a0.z,a0.w, a1.x,a1.y,a1.z,a1.w};
            const float u1[8] = {c0.x,c0.y,c0.z,c0.w, c1.x,c1.y,c1.z,c1.w};
            const float v0[8] = {e0.x,e0.y,e0.z,e0.w, e1.x,e1.y,e1.z,e1.w};
            const float v1[8] = {f0.x,f0.y,f0.z,f0.w, f1.x,f1.y,f1.z,f1.w};
            #pragma unroll
            for (int i = 0; i < 8; ++i) {
                *(unsigned int*)&sBu[bn + i][2 * kp] = pack_bf16x2(u0[i], u1[i]);
                *(unsigned int*)&sBv[bn + i][2 * kp] = pack_bf16x2(v0[i], v1[i]);
            }
            if (kk + TK + 1 < DIM) {   // prefetch next W slab -> global_prefetch_b8
                __builtin_prefetch((const void*)(Wu + (size_t)(kk + TK) * DIM + N0 + bn), 0, 1);
                __builtin_prefetch((const void*)(Wv + (size_t)(kk + TK) * DIM + N0 + bn), 0, 1);
            }
        }

        __builtin_amdgcn_s_wait_tensorcnt(0);   // own TDM slice landed in LDS
        __syncthreads();                        // all slices + B tiles visible

        // ---- compute: 16-row strip per wave, 8 N-tiles, both towers ----
        // 16-bit A/B fragment layout (ISA 7.12.2):
        //   lane<16 : K 0..7 (q[0]) and 16..23 (q[1]); lane>=16: K 8..15 / 24..31
        const int arow = wave * 16 + (lane & 15);
        const int kb   = (lane >> 4) * 8;

        FragBF aU, aV;
        aU.q[0] = *(const uint4*)&sAu[arow][kb];
        aU.q[1] = *(const uint4*)&sAu[arow][kb + 16];
        aV.q[0] = *(const uint4*)&sAv[arow][kb];
        aV.q[1] = *(const uint4*)&sAv[arow][kb + 16];

        // software pipeline: frags for tile j+1 load while WMMAs for tile j run
        FragBF bU, bV, bUn, bVn;
        {
            const int bcol = lane & 15;
            bU.q[0] = *(const uint4*)&sBu[bcol][kb];
            bU.q[1] = *(const uint4*)&sBu[bcol][kb + 16];
            bV.q[0] = *(const uint4*)&sBv[bcol][kb];
            bV.q[1] = *(const uint4*)&sBv[bcol][kb + 16];
        }
        #pragma unroll
        for (int j = 0; j < 8; ++j) {
            if (j < 7) {
                const int bcol = (j + 1) * 16 + (lane & 15);
                bUn.q[0] = *(const uint4*)&sBu[bcol][kb];
                bUn.q[1] = *(const uint4*)&sBu[bcol][kb + 16];
                bVn.q[0] = *(const uint4*)&sBv[bcol][kb];
                bVn.q[1] = *(const uint4*)&sBv[bcol][kb + 16];
            }
            accU[j] = __builtin_amdgcn_wmma_f32_16x16x32_bf16(
                          false, aU.v, false, bU.v, (short)0, accU[j], false, false);
            accV[j] = __builtin_amdgcn_wmma_f32_16x16x32_bf16(
                          false, aV.v, false, bV.v, (short)0, accV[j], false, false);
            bU = bUn;
            bV = bVn;
        }
    }

    // ---- epilogue: bias + LeakyReLU + elementwise product, reduce over N ----
    // C/D layout: VGPR r, lane<16 -> (M=r, N=lane); lane>=16 -> (M=r+8, N=lane-16)
    const int ncol = lane & 15;
    float bus[8], bvs[8];
    #pragma unroll
    for (int j = 0; j < 8; ++j) {
        bus[j] = bu[N0 + j * 16 + ncol];
        bvs[j] = bv[N0 + j * 16 + ncol];
    }
    #pragma unroll
    for (int r = 0; r < 8; ++r) {
        float sdot = 0.0f;
        #pragma unroll
        for (int j = 0; j < 8; ++j) {
            float hu = leaky(accU[j][r] + bus[j]);
            float hv = leaky(accV[j][r] + bvs[j]);
            sdot += hu * hv;
        }
        const int m = wave * 16 + ((lane >> 4) << 3) + r;
        __hip_atomic_fetch_add(&sRow[m], sdot,
                               __ATOMIC_RELAXED, __HIP_MEMORY_SCOPE_WORKGROUP); // ds_add_f32
    }
    __syncthreads();
    if (tid < TM) {
        __hip_atomic_fetch_add(&out[M0 + tid], sRow[tid],
                               __ATOMIC_RELAXED, __HIP_MEMORY_SCOPE_AGENT);     // global atomic add
    }
}

// ---------------------------------------------------------------------------
// Launch
// ---------------------------------------------------------------------------
extern "C" void kernel_launch(void* const* d_in, const int* in_sizes, int n_in,
                              void* d_out, int out_size, void* d_ws, size_t ws_size,
                              hipStream_t stream) {
    const float* u      = (const float*)d_in[0];
    const float* v      = (const float*)d_in[1];
    const float* ln_u_w = (const float*)d_in[2];
    const float* ln_u_b = (const float*)d_in[3];
    const float* ln_v_w = (const float*)d_in[4];
    const float* ln_v_b = (const float*)d_in[5];
    const float* W_u    = (const float*)d_in[6];
    const float* b_u    = (const float*)d_in[7];
    const float* W_v    = (const float*)d_in[8];
    const float* b_v    = (const float*)d_in[9];
    float* out = (float*)d_out;

    // workspace: LayerNormed bf16 activations (2 x 32 MB)
    unsigned short* Au = (unsigned short*)d_ws;
    unsigned short* Av = Au + (size_t)BATCH * DIM;

    zero_out_kernel<<<(BATCH + 255) / 256, 256, 0, stream>>>(out, BATCH);

    ln_rows_kernel<<<BATCH, 256, 0, stream>>>(u, ln_u_w, ln_u_b, Au);
    ln_rows_kernel<<<BATCH, 256, 0, stream>>>(v, ln_v_w, ln_v_b, Av);

    dim3 grid(BATCH / TM, DIM / TN);   // 64 x 16
    fused_gemm_dot_kernel<<<grid, 256, 0, stream>>>(Au, Av, W_u, W_v, b_u, b_v, out);
}
